// PerturbedTopK_6150393168241
// MI455X (gfx1250) — compile-verified, matrix operand order checked by reading, not run on hardware
//
#include <hip/hip_runtime.h>
#include <stdint.h>

// ---------------------------------------------------------------------------
// PerturbedTopK for MI455X (gfx1250, wave32)
//   x:(64,196) f32, noise:(64,500,196) f32 -> out:(64,49,196) f32
//   rank[d] = #{j : key[j] > key[d]}  (64-bit keys give exact top_k tie order)
//   in_top  = rank < 49 ; k = prefix count of in_top below d
//   rank row-sums computed on the matrix engine: comparison bits (f16 0/1)
//   x all-ones B via v_wmma_f32_16x16x32_f16, f32 accumulate (exact integers).
// ---------------------------------------------------------------------------

typedef __attribute__((ext_vector_type(16))) _Float16 v16h;
typedef __attribute__((ext_vector_type(8)))  float    v8f;

#define B_    64
#define N_    500
#define D_    196
#define K_    49
#define DP_   224          // padded length: 7 K-tiles * 32
#define NT_   13           // row tiles of 16 (13*16 = 208 >= 196)
#define NKT_  7            // K tiles of 32
#define SIGMA 0.05f
#define WAVES 8            // waves (rows) per 256-thread block

__global__ __launch_bounds__(256) void ptopk_main(const float* __restrict__ x,
                                                  const float* __restrict__ noise,
                                                  float* __restrict__ out) {
    __shared__ uint64_t keysh[WAVES][DP_];      // sortable keys per row
    __shared__ uint32_t intopw[WAVES][DP_ / 4]; // in-top bytes, packed u32

    const int wave = threadIdx.x >> 5;
    const int lane = threadIdx.x & 31;
    const int row  = blockIdx.x * WAVES + wave; // B_*N_ = 32000 = 4000*8 exact
    const int b    = row / N_;

    const float* xb = x + (size_t)b * D_;
    const float* nb = noise + (size_t)row * D_;
    uint64_t*  keys = keysh[wave];

    // --- build exact order keys: (orderedBits(v) << 8) | index; pad = 0 ----
    #pragma unroll
    for (int c = 0; c < DP_ / 32; ++c) {
        int d = c * 32 + lane;
        uint64_t key = 0;  // padding keys: 0 < any real key -> never "greater"
        if (d < D_) {
            float v = xb[d] + SIGMA * nb[d];
            uint32_t u  = __float_as_uint(v);
            uint32_t ou = (u & 0x80000000u) ? ~u : (u | 0x80000000u);
            key = ((uint64_t)ou << 8) | (uint32_t)d;
        }
        keys[d] = key;
    }
    __syncthreads();

    // --- rank via WMMA: A tile row m covered by lanes m and m+16 -----------
    uint64_t km[NT_];
    #pragma unroll
    for (int t = 0; t < NT_; ++t) km[t] = keys[t * 16 + (lane & 15)];

    v16h bones;
    #pragma unroll
    for (int i = 0; i < 16; ++i) bones[i] = (_Float16)1.0f;

    const v8f zacc = {};
    v8f acc[NT_];
    #pragma unroll
    for (int t = 0; t < NT_; ++t) acc[t] = zacc;

    const int half16 = (lane >> 4) * 16;   // lane<16 -> j+0..15, lane>=16 -> j+16..31
    for (int tk = 0; tk < NKT_; ++tk) {
        uint64_t kj[16];
        #pragma unroll
        for (int i = 0; i < 16; ++i) kj[i] = keys[tk * 32 + half16 + i];
        #pragma unroll
        for (int t = 0; t < NT_; ++t) {
            v16h a;
            #pragma unroll
            for (int i = 0; i < 16; ++i)
                a[i] = (kj[i] > km[t]) ? (_Float16)1.0f : (_Float16)0.0f;
            // D = A(16x32 cmp bits) * B(ones) + C : every D column = row sum
            acc[t] = __builtin_amdgcn_wmma_f32_16x16x32_f16(
                false, a, false, bones, (short)0, acc[t], false, false);
        }
    }

    // --- lanes 0 / 16 hold rank[tile*16 + 0..7] / [+8..15]; pack in-top ----
    if ((lane & 15) == 0) {
        const int half = lane >> 4;
        #pragma unroll
        for (int t = 0; t < NT_; ++t) {
            uint32_t w0 = 0, w1 = 0;
            #pragma unroll
            for (int r = 0; r < 4; ++r) {
                w0 |= (acc[t][r]     < (float)K_ ? 1u : 0u) << (8 * r);
                w1 |= (acc[t][r + 4] < (float)K_ ? 1u : 0u) << (8 * r);
            }
            intopw[wave][t * 4 + half * 2 + 0] = w0;
            intopw[wave][t * 4 + half * 2 + 1] = w1;
        }
    }
    __syncthreads();

    // --- ballot prefix scan -> sorted position k; scatter +1 ---------------
    const uint8_t* ip = (const uint8_t*)intopw[wave];
    int base = 0;
    #pragma unroll
    for (int c = 0; c < DP_ / 32; ++c) {
        int  d    = c * 32 + lane;
        bool pred = false;
        if (d < D_) pred = (ip[d] != 0);
        unsigned mask = (unsigned)__ballot(pred);
        if (pred) {
            int k = base + __popc(mask & ((1u << lane) - 1u));
            atomicAdd(out + ((size_t)b * K_ + k) * D_ + d, 1.0f);
        }
        base += __popc(mask);
    }
}

__global__ void ptopk_zero(float* __restrict__ out, int total) {
    int i = blockIdx.x * 256 + threadIdx.x;
    if (i < total) out[i] = 0.0f;
}

__global__ void ptopk_scale(float* __restrict__ out, int total) {
    int i = blockIdx.x * 256 + threadIdx.x;
    if (i < total) out[i] = out[i] / (float)N_;   // exact match to reference /n
}

extern "C" void kernel_launch(void* const* d_in, const int* in_sizes, int n_in,
                              void* d_out, int out_size, void* d_ws, size_t ws_size,
                              hipStream_t stream) {
    (void)in_sizes; (void)n_in; (void)d_ws; (void)ws_size;
    const float* x     = (const float*)d_in[0];
    const float* noise = (const float*)d_in[1];
    float*       out   = (float*)d_out;

    const int total = out_size;                 // 64*49*196 = 614656
    const int zb    = (total + 255) / 256;

    ptopk_zero <<<zb, 256, 0, stream>>>(out, total);
    ptopk_main <<<(B_ * N_) / WAVES, 256, 0, stream>>>(x, noise, out);
    ptopk_scale<<<zb, 256, 0, stream>>>(out, total);
}
